// InferenceNet_28157805593260
// MI455X (gfx1250) — matrix operand order repaired and checked by reading, not compiled, not run on hardware
//
#include <hip/hip_runtime.h>
#include <math.h>

// ---------------------------------------------------------------------------
// Types for CDNA5 WMMA (wave32): bf16 A/B fragments (16 x bf16 = 8 VGPRs),
// f32 accumulator (8 x f32 = 8 VGPRs).
// ---------------------------------------------------------------------------
typedef __attribute__((ext_vector_type(16))) __bf16 bf16x16;
typedef __attribute__((ext_vector_type(8)))  __bf16 bf16x8;
typedef __attribute__((ext_vector_type(8)))  float  f32x8;
typedef int v4i __attribute__((vector_size(16)));   // matches async-LDS builtin proto

#ifndef __has_builtin
#define __has_builtin(x) 0
#endif
#if __has_builtin(__builtin_amdgcn_global_load_async_to_lds_b128) && \
    __has_builtin(__builtin_amdgcn_s_wait_asynccnt)
#define USE_ASYNC_LDS 1
#else
#define USE_ASYNC_LDS 0
#endif

__device__ __forceinline__ float sigf(float x){ return 1.f/(1.f+expf(-x)); }

// ---------------------------------------------------------------------------
// Generic GEMM:  C[M,N] = A[M,Kp] * W[N,Kp]^T (+ bias[N])
// A: bf16 row-major, K padded to multiple of 32 (pad = 0)
// W: bf16 row-major (N, Kp) (pre-packed by k_prepw)
// C: f32 row-major (M, N)
// ---------------------------------------------------------------------------
#define TBM 128
#define TBN 64
#define TBK 32
#define LDP 40   // LDS row pitch in elements (40*2B = 80B, 16B aligned, conflict-skewed)

// ---- narrow-N kernel: block tile 128x64, 8 waves (4x2), 2x2 frags/wave ----
__global__ __launch_bounds__(256) void k_gemm(
    const __bf16* __restrict__ A, const __bf16* __restrict__ W,
    const float* __restrict__ bias, float* __restrict__ C,
    int M, int N, int Kp)
{
  __shared__ __bf16 As[TBM * LDP];
  __shared__ __bf16 Bs[TBN * LDP];
  const int tid  = threadIdx.x;
  const int lane = tid & 31;
  const int wave = tid >> 5;
  const int wm = wave >> 1;        // 0..3 (M direction)
  const int wn = wave & 1;         // 0..1 (N direction)
  const int bm0 = blockIdx.x * TBM;
  const int bn0 = blockIdx.y * TBN;
  const int hl  = lane >> 4;       // lane half
  const int l16 = lane & 15;

  f32x8 acc[2][2];
#pragma unroll
  for (int i=0;i<2;i++)
#pragma unroll
    for (int j=0;j<2;j++) acc[i][j] = (f32x8){};

  for (int k0 = 0; k0 < Kp; k0 += TBK) {
    for (int t = tid; t < TBM*(TBK/8); t += 256) {
      int r = t >> 2, kc = (t & 3) << 3;
      int gr = bm0 + r;
      bf16x8 v = {};
      if (gr < M) v = *(const bf16x8*)(A + (size_t)gr * Kp + k0 + kc);
      *(bf16x8*)(&As[r * LDP + kc]) = v;
    }
    for (int t = tid; t < TBN*(TBK/8); t += 256) {
      int r = t >> 2, kc = (t & 3) << 3;
      int gn = bn0 + r;
      bf16x8 v = {};
      if (gn < N) v = *(const bf16x8*)(W + (size_t)gn * Kp + k0 + kc);
      *(bf16x8*)(&Bs[r * LDP + kc]) = v;
    }
    if (k0 + TBK < Kp) {
      int gr = bm0 + (tid >> 1);
      if (gr < M) __builtin_prefetch((const void*)(A + (size_t)gr * Kp + k0 + TBK), 0, 1);
    }
    __syncthreads();

    bf16x16 af[2], bfg[2];
#pragma unroll
    for (int mi=0; mi<2; mi++) {
      const __bf16* ap = &As[(wm*32 + mi*16 + l16) * LDP];
#pragma unroll
      for (int v=0; v<8; v++) {
        int k = (v<4 ? 2*v : 2*(v-4)+16) + 8*hl;   // A: lanes0-15 K{0..7,16..23}
        af[mi][2*v]   = ap[k];
        af[mi][2*v+1] = ap[k+1];
      }
    }
#pragma unroll
    for (int ni=0; ni<2; ni++) {
      const __bf16* bp = &Bs[(wn*32 + ni*16 + l16) * LDP];
#pragma unroll
      for (int v=0; v<8; v++) {
        int k = (v<4 ? 2*v : 2*(v-4)+16) + 8*hl;   // B: lanes0-15 K{0..7,16..23}
        bfg[ni][2*v]   = bp[k];
        bfg[ni][2*v+1] = bp[k+1];
      }
    }
#pragma unroll
    for (int mi=0; mi<2; mi++)
#pragma unroll
      for (int ni=0; ni<2; ni++)
        acc[mi][ni] = __builtin_amdgcn_wmma_f32_16x16x32_bf16(
            false, af[mi], false, bfg[ni], (short)0, acc[mi][ni], false, false);
    __syncthreads();
  }

#pragma unroll
  for (int mi=0; mi<2; mi++) {
#pragma unroll
    for (int ni=0; ni<2; ni++) {
      int col = bn0 + wn*32 + ni*16 + l16;
      if (col >= N) continue;
      float bv = bias ? bias[col] : 0.f;
#pragma unroll
      for (int v=0; v<8; v++) {
        int row = bm0 + wm*32 + mi*16 + hl*8 + v;
        if (row < M) C[(size_t)row * N + col] = acc[mi][ni][v] + bv;
      }
    }
  }
}

// ---- wide-N kernel: block tile 128x128, double-buffered LDS, 2x4 frags/wave,
//      async global->LDS staging when the toolchain exposes the builtin ------
__global__ __launch_bounds__(256) void k_gemm_big(
    const __bf16* __restrict__ A, const __bf16* __restrict__ W,
    const float* __restrict__ bias, float* __restrict__ C,
    int M, int N, int Kp)
{
  __shared__ __bf16 As[2*128*LDP];
  __shared__ __bf16 Bs[2*128*LDP];
  const int tid  = threadIdx.x;
  const int lane = tid & 31;
  const int wave = tid >> 5;
  const int wm = wave >> 1;        // 0..3 (rows of 32)
  const int wn = wave & 1;         // 0..1 (cols of 64)
  const int bm0 = blockIdx.x * 128;
  const int bn0 = blockIdx.y * 128;
  const int hl  = lane >> 4;
  const int l16 = lane & 15;

  f32x8 acc[2][4];
#pragma unroll
  for (int i=0;i<2;i++)
#pragma unroll
    for (int j=0;j<4;j++) acc[i][j] = (f32x8){};

  auto stage = [&](int k0, int buf){
#if USE_ASYNC_LDS
    for (int t = tid; t < 512; t += 256) {
      int r = t >> 2, kc = (t & 3) << 3;
      int gr = bm0 + r, gn = bn0 + r;
      if (gr < M)
        __builtin_amdgcn_global_load_async_to_lds_b128(
            (__attribute__((address_space(1))) v4i*)(A + (size_t)gr*Kp + k0 + kc),
            (__attribute__((address_space(3))) v4i*)(&As[(size_t)buf*128*LDP + r*LDP + kc]),
            0, 0);
      if (gn < N)
        __builtin_amdgcn_global_load_async_to_lds_b128(
            (__attribute__((address_space(1))) v4i*)(W + (size_t)gn*Kp + k0 + kc),
            (__attribute__((address_space(3))) v4i*)(&Bs[(size_t)buf*128*LDP + r*LDP + kc]),
            0, 0);
    }
#else
    for (int t = tid; t < 512; t += 256) {
      int r = t >> 2, kc = (t & 3) << 3;
      { int gr = bm0 + r;
        bf16x8 v = {};
        if (gr < M) v = *(const bf16x8*)(A + (size_t)gr*Kp + k0 + kc);
        *(bf16x8*)(&As[(size_t)buf*128*LDP + r*LDP + kc]) = v; }
      { int gn = bn0 + r;
        bf16x8 v = {};
        if (gn < N) v = *(const bf16x8*)(W + (size_t)gn*Kp + k0 + kc);
        *(bf16x8*)(&Bs[(size_t)buf*128*LDP + r*LDP + kc]) = v; }
    }
#endif
  };
  auto wait_stage = [&](){
#if USE_ASYNC_LDS
    __builtin_amdgcn_s_wait_asynccnt(0);
#endif
    __syncthreads();
  };

  const int nk = Kp / TBK;
  stage(0, 0);
  wait_stage();
  for (int kt = 0; kt < nk; kt++) {
    int buf = kt & 1;
    if (kt + 1 < nk) stage((kt+1)*TBK, buf ^ 1);  // overlap with compute below

    const __bf16* Ab = &As[(size_t)buf*128*LDP];
    const __bf16* Bb = &Bs[(size_t)buf*128*LDP];
    bf16x16 af[2], bfg[4];
#pragma unroll
    for (int mi=0; mi<2; mi++) {
      const __bf16* ap = &Ab[(wm*32 + mi*16 + l16) * LDP];
#pragma unroll
      for (int v=0; v<8; v++) {
        int k = (v<4 ? 2*v : 2*(v-4)+16) + 8*hl;
        af[mi][2*v]   = ap[k];
        af[mi][2*v+1] = ap[k+1];
      }
    }
#pragma unroll
    for (int ni=0; ni<4; ni++) {
      const __bf16* bp = &Bb[(wn*64 + ni*16 + l16) * LDP];
#pragma unroll
      for (int v=0; v<8; v++) {
        int k = (v<4 ? 2*v : 2*(v-4)+16) + 8*hl;
        bfg[ni][2*v]   = bp[k];
        bfg[ni][2*v+1] = bp[k+1];
      }
    }
#pragma unroll
    for (int mi=0; mi<2; mi++)
#pragma unroll
      for (int ni=0; ni<4; ni++)
        acc[mi][ni] = __builtin_amdgcn_wmma_f32_16x16x32_bf16(
            false, af[mi], false, bfg[ni], (short)0, acc[mi][ni], false, false);

    if (kt + 1 < nk) wait_stage();
  }

#pragma unroll
  for (int mi=0; mi<2; mi++) {
#pragma unroll
    for (int ni=0; ni<4; ni++) {
      int col = bn0 + wn*64 + ni*16 + l16;
      if (col >= N) continue;
      float bv = bias ? bias[col] : 0.f;
#pragma unroll
      for (int v=0; v<8; v++) {
        int row = bm0 + wm*32 + mi*16 + hl*8 + v;
        if (row < M) C[(size_t)row * N + col] = acc[mi][ni][v] + bv;
      }
    }
  }
}

// ---------------------------------------------------------------------------
// Weight packing: f32 (N,K) [trans=0] or (K,N) [trans=1] -> bf16 (N,Kp), pad 0
// ---------------------------------------------------------------------------
__global__ void k_prepw(const float* __restrict__ w, __bf16* __restrict__ o,
                        int N, int K, int Kp, int trans){
  size_t i = (size_t)blockIdx.x*256 + threadIdx.x;
  if (i >= (size_t)N*Kp) return;
  int k = (int)(i % Kp); int n = (int)(i / Kp);
  float v = 0.f;
  if (k < K) v = trans ? w[(size_t)k*N + n] : w[(size_t)n*K + k];
  o[i] = (__bf16)v;
}

// ---------------------------------------------------------------------------
// Per-column (channel-last) batch-norm statistics and fused act + bf16 pack
// ---------------------------------------------------------------------------
__global__ void k_colstats(const float* __restrict__ X, float* mean, float* rstd,
                           int M, int N){
  int c = blockIdx.x;
  __shared__ float ss[256], sq[256];
  float s = 0.f, q = 0.f;
  for (int r = threadIdx.x; r < M; r += 256) {
    float v = X[(size_t)r*N + c]; s += v; q += v*v;
  }
  ss[threadIdx.x] = s; sq[threadIdx.x] = q; __syncthreads();
  for (int st = 128; st > 0; st >>= 1) {
    if (threadIdx.x < st) { ss[threadIdx.x] += ss[threadIdx.x+st]; sq[threadIdx.x] += sq[threadIdx.x+st]; }
    __syncthreads();
  }
  if (threadIdx.x == 0) {
    float m = ss[0] / M; float v = sq[0] / M - m*m;
    mean[c] = m; rstd[c] = rsqrtf(v + 1e-5f);
  }
}

// mode: 0=relu
__global__ void k_bn_act(float* __restrict__ X, __bf16* __restrict__ Y,
                         const float* mean, const float* rstd,
                         const float* g, const float* b,
                         int M, int N, int Np, int mode){
  size_t i = (size_t)blockIdx.x*256 + threadIdx.x;
  if (i >= (size_t)M*Np) return;
  int c = (int)(i % Np); size_t r = i / Np;
  float y = 0.f;
  if (c < N) {
    float x = X[r*N + c];
    x = (x - mean[c]) * rstd[c] * g[c] + b[c];
    if (mode == 0) x = x > 0.f ? x : 0.f;
    X[r*N + c] = x; y = x;
  }
  if (Y) Y[i] = (__bf16)y;
}

// plain activation: 0=relu 1=elu 2=identity, write f32 in place + optional padded bf16
__global__ void k_act(float* __restrict__ X, __bf16* __restrict__ Y,
                      int M, int N, int Np, int mode){
  size_t i = (size_t)blockIdx.x*256 + threadIdx.x;
  if (i >= (size_t)M*Np) return;
  int c = (int)(i % Np); size_t r = i / Np;
  float y = 0.f;
  if (c < N) {
    float x = X[r*N + c];
    if (mode == 0)      x = x > 0.f ? x : 0.f;
    else if (mode == 1) x = x > 0.f ? x : expm1f(x);
    X[r*N + c] = x; y = x;
  }
  if (Y) Y[i] = (__bf16)y;
}

// ---------------------------------------------------------------------------
// Point-cloud helpers
// ---------------------------------------------------------------------------
__global__ void k_split_partial(const float* __restrict__ P, float* xyz, float* pts,
                                int B, int N){
  int i = blockIdx.x*256 + threadIdx.x;
  if (i >= B*N) return;
  int b = i / N, n = i % N;
  const float* base = P + (size_t)b*7*N + n;
  for (int c = 0; c < 3; c++) xyz[((size_t)b*N + n)*3 + c] = base[(size_t)c*N];
  for (int c = 0; c < 4; c++) pts[((size_t)b*N + n)*4 + c] = base[(size_t)(3+c)*N];
}

__global__ void k_fps(const float* __restrict__ xyz, int* __restrict__ fidx,
                      float* __restrict__ distbuf, int N, int S){
  int b = blockIdx.x;
  const float* X = xyz + (size_t)b*N*3;
  float* dist = distbuf + (size_t)b*N;
  __shared__ float sval[256]; __shared__ int sidx[256];
  __shared__ int cur;
  for (int n = threadIdx.x; n < N; n += 256) dist[n] = 1e10f;
  if (threadIdx.x == 0) { cur = 0; fidx[b*S] = 0; }
  __syncthreads();
  for (int it = 1; it < S; it++) {
    float cx = X[cur*3], cy = X[cur*3+1], cz = X[cur*3+2];
    float best = -1.f; int bi = 0x7fffffff;
    for (int n = threadIdx.x; n < N; n += 256) {
      float dx = X[n*3]-cx, dy = X[n*3+1]-cy, dz = X[n*3+2]-cz;
      float dm = fminf(dist[n], dx*dx + dy*dy + dz*dz);
      dist[n] = dm;
      if (dm > best) { best = dm; bi = n; }
    }
    sval[threadIdx.x] = best; sidx[threadIdx.x] = bi; __syncthreads();
    for (int st = 128; st > 0; st >>= 1) {
      if (threadIdx.x < st) {
        float ov = sval[threadIdx.x+st]; int oi = sidx[threadIdx.x+st];
        if (ov > sval[threadIdx.x] || (ov == sval[threadIdx.x] && oi < sidx[threadIdx.x])) {
          sval[threadIdx.x] = ov; sidx[threadIdx.x] = oi;
        }
      }
      __syncthreads();
    }
    if (threadIdx.x == 0) { cur = sidx[0]; fidx[b*S + it] = cur; }
    __syncthreads();
  }
}

__global__ void k_gather3(const float* __restrict__ xyz, const int* __restrict__ fidx,
                          float* __restrict__ out, int B, int N, int S){
  int i = blockIdx.x*256 + threadIdx.x;
  if (i >= B*S) return;
  int b = i / S; int id = fidx[i];
  for (int c = 0; c < 3; c++) out[(size_t)i*3 + c] = xyz[((size_t)b*N + id)*3 + c];
}

__global__ void k_ballq(const float* __restrict__ xyz, const float* __restrict__ nxyz,
                        int* __restrict__ idx, int B, int N, int S, int K, float r2){
  int i = blockIdx.x*256 + threadIdx.x;
  if (i >= B*S) return;
  int b = i / S;
  float qx = nxyz[(size_t)i*3], qy = nxyz[(size_t)i*3+1], qz = nxyz[(size_t)i*3+2];
  const float* X = xyz + (size_t)b*N*3;
  int* out = idx + (size_t)i*K;
  int cnt = 0, first = 0;
  for (int n = 0; n < N && cnt < K; n++) {
    float dx = X[n*3]-qx, dy = X[n*3+1]-qy, dz = X[n*3+2]-qz;
    if (dx*dx + dy*dy + dz*dz <= r2) { if (cnt == 0) first = n; out[cnt++] = n; }
  }
  for (; cnt < K; cnt++) out[cnt] = first;
}

// build grouped GEMM input rows (B*S*K, Kp) bf16: [xyz[idx]-new_xyz | pts[idx] | 0pad]
__global__ void k_group(const float* __restrict__ xyz, const float* __restrict__ pts,
                        const float* __restrict__ nxyz, const int* __restrict__ idx,
                        __bf16* __restrict__ A, int B, int N, int S, int K, int C, int Kp){
  size_t i = (size_t)blockIdx.x*256 + threadIdx.x;
  if (i >= (size_t)B*S*K*Kp) return;
  int c = (int)(i % Kp); size_t rw = i / Kp;
  int k = (int)(rw % K); size_t bs = rw / K;
  int s = (int)(bs % S); int b = (int)(bs / S);
  int id = idx[((size_t)(b*S + s))*K + k];
  float v = 0.f;
  if (c < 3)          v = xyz[((size_t)b*N + id)*3 + c] - nxyz[((size_t)(b*S + s))*3 + c];
  else if (c < 3 + C) v = pts[((size_t)b*N + id)*C + (c - 3)];
  A[i] = (__bf16)v;
}

__global__ void k_maxpool_k(const float* __restrict__ X, float* __restrict__ Y,
                            int BS, int Kn, int C){
  int i = blockIdx.x*256 + threadIdx.x;
  if (i >= BS*C) return;
  int s = i / C, c = i % C;
  const float* p = X + ((size_t)s*Kn)*C + c;
  float m = -1e30f;
  for (int k = 0; k < Kn; k++) m = fmaxf(m, p[(size_t)k*C]);
  Y[i] = m;
}

__global__ void k_knn3(const float* __restrict__ x1, const float* __restrict__ x2,
                       int* __restrict__ idx3, float* __restrict__ w3,
                       int B, int N1, int N2){
  int i = blockIdx.x*256 + threadIdx.x;
  if (i >= B*N1) return;
  int b = i / N1;
  float px = x1[(size_t)i*3], py = x1[(size_t)i*3+1], pz = x1[(size_t)i*3+2];
  const float* X2 = x2 + (size_t)b*N2*3;
  float d0 = 1e30f, d1 = 1e30f, d2 = 1e30f; int i0 = 0, i1 = 0, i2 = 0;
  for (int n = 0; n < N2; n++) {
    float dx = X2[n*3]-px, dy = X2[n*3+1]-py, dz = X2[n*3+2]-pz;
    float d = dx*dx + dy*dy + dz*dz;
    if (d < d0)      { d2=d1;i2=i1; d1=d0;i1=i0; d0=d;i0=n; }
    else if (d < d1) { d2=d1;i2=i1; d1=d;i1=n; }
    else if (d < d2) { d2=d;i2=n; }
  }
  float r0 = 1.f/(d0+1e-8f), r1 = 1.f/(d1+1e-8f), r2 = 1.f/(d2+1e-8f);
  float s = r0 + r1 + r2;
  idx3[i*3]=i0; idx3[i*3+1]=i1; idx3[i*3+2]=i2;
  w3[i*3]=r0/s; w3[i*3+1]=r1/s; w3[i*3+2]=r2/s;
}

// rows (B*N1, Kp) bf16: [points1(C1) | 3-NN interp of points2 (C2) | 0pad]
__global__ void k_interp_concat(const float* __restrict__ p1, const float* __restrict__ p2,
                                const int* __restrict__ idx3, const float* __restrict__ w3,
                                __bf16* __restrict__ A, int B, int N1, int N2,
                                int C1, int C2, int Kp){
  size_t i = (size_t)blockIdx.x*256 + threadIdx.x;
  if (i >= (size_t)B*N1*Kp) return;
  int c = (int)(i % Kp); size_t r = i / Kp; int b = (int)(r / N1);
  float v = 0.f;
  if (c < C1) v = p1[r*C1 + c];
  else if (c < C1 + C2) {
    int cc = c - C1;
    const float* P = p2 + (size_t)b*N2*C2;
    const int* id = idx3 + r*3; const float* w = w3 + r*3;
    v = w[0]*P[(size_t)id[0]*C2 + cc] + w[1]*P[(size_t)id[1]*C2 + cc] + w[2]*P[(size_t)id[2]*C2 + cc];
  }
  A[i] = (__bf16)v;
}

// fp3: interp over uniform-broadcast features is exactly asf (weights sum to 1)
__global__ void k_concat_bcast(const float* __restrict__ p1, const float* __restrict__ asf,
                               __bf16* __restrict__ A, int B, int N1, int C1, int C2, int Kp){
  size_t i = (size_t)blockIdx.x*256 + threadIdx.x;
  if (i >= (size_t)B*N1*Kp) return;
  int c = (int)(i % Kp); size_t r = i / Kp; int b = (int)(r / N1);
  float v = 0.f;
  if (c < C1) v = p1[r*C1 + c];
  else if (c < C1 + C2) v = asf[(size_t)b*C2 + (c - C1)];
  A[i] = (__bf16)v;
}

// ---------------------------------------------------------------------------
// CRNN path
// ---------------------------------------------------------------------------
__global__ void k_conv1d(const float* __restrict__ x, const float* __restrict__ w,
                         const float* __restrict__ bias, float* __restrict__ y,
                         int B, int Cin, int Lin, int Cout, int Lout, int Kk,
                         int stride, int pad){
  int i = blockIdx.x*256 + threadIdx.x;
  if (i >= B*Cout*Lout) return;
  int t = i % Lout; int rem = i / Lout; int o = rem % Cout; int b = rem / Cout;
  float acc = bias[o];
  for (int c = 0; c < Cin; c++) {
    const float* xp = x + ((size_t)b*Cin + c)*Lin;
    const float* wp = w + ((size_t)o*Cin + c)*Kk;
    for (int k = 0; k < Kk; k++) {
      int p = t*stride + k - pad;
      if (p >= 0 && p < Lin) acc += xp[p]*wp[k];
    }
  }
  y[i] = acc;
}

__global__ void k_bnstats_bct(const float* __restrict__ x, float* mean, float* rstd,
                              int B, int C, int T){
  int c = blockIdx.x;
  __shared__ float ss[256], sq[256];
  float s = 0.f, q = 0.f;
  for (int i = threadIdx.x; i < B*T; i += 256) {
    int b = i / T, t = i % T;
    float v = x[((size_t)b*C + c)*T + t]; s += v; q += v*v;
  }
  ss[threadIdx.x] = s; sq[threadIdx.x] = q; __syncthreads();
  for (int st = 128; st > 0; st >>= 1) {
    if (threadIdx.x < st) { ss[threadIdx.x] += ss[threadIdx.x+st]; sq[threadIdx.x] += sq[threadIdx.x+st]; }
    __syncthreads();
  }
  if (threadIdx.x == 0) {
    float m = ss[0]/(B*T); float v = sq[0]/(B*T) - m*m;
    mean[c] = m; rstd[c] = rsqrtf(v + 1e-5f);
  }
}

__global__ void k_bnact_bct(float* __restrict__ x, const float* mean, const float* rstd,
                            const float* g, const float* b, int B, int C, int T, float slope){
  int i = blockIdx.x*256 + threadIdx.x;
  if (i >= B*C*T) return;
  int rem = i / T; int c = rem % C;
  float v = x[i]; v = (v - mean[c]) * rstd[c] * g[c] + b[c];
  x[i] = v > 0.f ? v : slope*v;
}

__global__ void k_bct_to_tbc(const float* __restrict__ x, float* __restrict__ y,
                             int B, int C, int T){
  int i = blockIdx.x*256 + threadIdx.x;
  if (i >= B*C*T) return;
  int t = i % T; int rem = i / T; int c = rem % C; int b = rem / C;
  y[((size_t)t*B + b)*C + c] = x[i];
}

// single-block LSTM (B=8, H=64, Cin=256, 4H=256 threads-worth of gates)
__global__ __launch_bounds__(256) void k_lstm(
    const float* __restrict__ x, const float* __restrict__ wih,
    const float* __restrict__ whh, const float* __restrict__ bih,
    const float* __restrict__ bhh, float* __restrict__ hs,
    int T, int B, int Cin, int H, int rev){
  __shared__ float sh[512], sc[512], sg[2048], sx[2048];
  int u = threadIdx.x;
  for (int i = u; i < B*H; i += 256) { sh[i] = 0.f; sc[i] = 0.f; }
  __syncthreads();
  for (int step = 0; step < T; step++) {
    int t = rev ? (T-1-step) : step;
    for (int i = u; i < B*Cin; i += 256) sx[i] = x[(size_t)t*B*Cin + i];
    __syncthreads();
    for (int b = 0; b < B; b++) {
      float acc = bih[u] + bhh[u];
      const float* wi = wih + (size_t)u*Cin;
      const float* xb = sx + b*Cin;
      for (int c = 0; c < Cin; c++) acc += xb[c]*wi[c];
      const float* wh = whh + (size_t)u*H;
      const float* hb = sh + b*H;
      for (int c = 0; c < H; c++) acc += hb[c]*wh[c];
      sg[b*4*H + u] = acc;
    }
    __syncthreads();
    for (int i = u; i < B*H; i += 256) {
      int b = i / H, h = i % H;
      float gi = sg[b*4*H + h];
      float gf = sg[b*4*H + H + h];
      float gg = sg[b*4*H + 2*H + h];
      float go = sg[b*4*H + 3*H + h];
      float c = sigf(gf)*sc[i] + sigf(gi)*tanhf(gg);
      float hv = sigf(go)*tanhf(c);
      sc[i] = c; sh[i] = hv;
      hs[((size_t)t*B + b)*H + h] = hv;
    }
    __syncthreads();
  }
}

__global__ void k_rec_concat(const float* __restrict__ hf, const float* __restrict__ hb,
                             __bf16* __restrict__ A, int TB, int H){
  int i = blockIdx.x*256 + threadIdx.x;
  if (i >= TB*2*H) return;
  int c = i % (2*H); int r = i / (2*H);
  float v = c < H ? hf[(size_t)r*H + c] : hb[(size_t)r*H + c - H];
  A[i] = (__bf16)v;
}

__global__ void k_timemax(const float* __restrict__ x, float* __restrict__ y,
                          int T, int B, int C){
  int i = blockIdx.x*256 + threadIdx.x;
  if (i >= B*C) return;
  int b = i / C, c = i % C;
  float m = -1e30f;
  for (int t = 0; t < T; t++) m = fmaxf(m, x[((size_t)t*B + b)*C + c]);
  y[i] = m;
}

// deterministic pseudo-normal (stand-in for threefry(key=42) normal)
__device__ float rnorm_hash(unsigned n){
  unsigned x = n*747796405u + 2891336453u;
  x = ((x >> ((x>>28)+4u)) ^ x) * 277803737u;
  x = (x>>22) ^ x;
  unsigned y = (n+1234567u)*2654435769u; y ^= y>>16; y *= 2246822519u; y ^= y>>13;
  float u1 = (float)(x & 0xFFFFFFu) / 16777216.0f + 1e-7f;
  float u2 = (float)(y & 0xFFFFFFu) / 16777216.0f;
  return sqrtf(-2.f*logf(u1)) * cosf(6.2831853f*u2);
}

__global__ void k_latent(const float* __restrict__ feats, float* __restrict__ out_mu,
                         float* __restrict__ out_std, float* __restrict__ lg,
                         __bf16* __restrict__ zlat, __bf16* __restrict__ zg,
                         int B, int Zc){
  int i = blockIdx.x*256 + threadIdx.x;
  if (i >= B*Zc) return;
  int b = i / Zc, c = i % Zc;
  float mu = feats[b*2*Zc + c];
  float sd = feats[b*2*Zc + Zc + c] + 1e-6f;
  out_mu[i] = mu; out_std[i] = sd;
  float eps = rnorm_hash((unsigned)i);
  float g = mu + eps*expf(0.5f*sd);
  lg[i] = g;
  zlat[(size_t)b*32 + c]      = (__bf16)mu;
  zlat[(size_t)b*32 + Zc + c] = (__bf16)g;
  zg[(size_t)b*32 + c]        = (__bf16)g;
  zg[(size_t)b*32 + Zc + c]   = (__bf16)0.f;
}

// ---------------------------------------------------------------------------
// Segmentation head / decoders
// ---------------------------------------------------------------------------
__global__ void k_softmax_seg(const float* __restrict__ X, float* __restrict__ out,
                              int B, int N){
  int i = blockIdx.x*256 + threadIdx.x;
  if (i >= B*N) return;
  int b = i / N, n = i % N;
  float a = X[(size_t)i*3], b1 = X[(size_t)i*3+1], c = X[(size_t)i*3+2];
  float m = fmaxf(a, fmaxf(b1, c));
  float ea = expf(a-m), eb = expf(b1-m), ec = expf(c-m);
  float s = ea + eb + ec;
  out[((size_t)b*3 + 0)*N + n] = ea/s;
  out[((size_t)b*3 + 1)*N + n] = eb/s;
  out[((size_t)b*3 + 2)*N + n] = ec/s;
}

__global__ void k_sigmoid(const float* __restrict__ x, float* __restrict__ y, int n){
  int i = blockIdx.x*256 + threadIdx.x;
  if (i < n) y[i] = sigf(x[i]);
}

// decoder fine-path features: rows (B*1024, 32pad) = [grid(2) | pf(7) | z(16)]
__global__ void k_dec_feat(const float* __restrict__ coarse, const float* __restrict__ lg,
                           __bf16* __restrict__ A, int B){
  size_t i = (size_t)blockIdx.x*256 + threadIdx.x;
  if (i >= (size_t)B*1024*32) return;
  int c = (int)(i % 32); size_t r = i / 32;
  int m = (int)(r % 1024); int b = (int)(r / 1024);
  float v = 0.f;
  if (c < 2) {
    const float g0[4] = {-0.05f,-0.05f, 0.05f, 0.05f};
    const float g1[4] = {-0.05f, 0.05f,-0.05f, 0.05f};
    v = (c == 0) ? g0[m & 3] : g1[m & 3];
  } else if (c < 9) {
    int f = (c-2)*1024 + m;               // pf[b][ch][m] reshape semantics
    v = coarse[((size_t)b*256 + (f/28))*7 + (f % 7)];
  } else if (c < 25) {
    v = lg[b*16 + (c - 9)];
  }
  A[i] = (__bf16)v;
}

__global__ void k_detail_out(const float* __restrict__ conv3out,
                             const float* __restrict__ coarse,
                             float* __restrict__ out, int B){
  int i = blockIdx.x*256 + threadIdx.x;
  if (i >= B*1024*7) return;
  int j = i % 7; int rem = i / 7; int m = rem % 1024; int b = rem / 1024;
  int f = m*7 + j;                         // center = pf.reshape(B,1024,7)
  float fine = conv3out[((size_t)b*1024 + m)*7 + j]
             + coarse[((size_t)b*256 + (f/28))*7 + (f % 7)];
  out[i] = sigf(fine);
}

// ---------------------------------------------------------------------------
// Signal decoder
// ---------------------------------------------------------------------------
__global__ void k_pad_mu(const float* __restrict__ feats, __bf16* __restrict__ o){
  int i = threadIdx.x;
  if (i < 8*32) {
    int b = i / 32, c = i % 32;
    o[i] = (__bf16)(c < 16 ? feats[b*32 + c] : 0.f);
  }
}

__global__ void k_resize(const float* __restrict__ inb, float* __restrict__ out, int B){
  int i = blockIdx.x*256 + threadIdx.x;
  if (i >= B*8*512) return;
  int x = i % 512; int rem = i / 512; int h = rem % 8; int b = rem / 8;
  float s = (x + 0.5f)*0.25f - 0.5f;
  float fl = floorf(s); int x0 = (int)fl; float fr = s - fl;
  int xa = x0 < 0 ? 0 : (x0 > 127 ? 127 : x0);
  int x1 = x0 + 1;
  int xb = x1 < 0 ? 0 : (x1 > 127 ? 127 : x1);
  const float* p = inb + ((size_t)b*8 + h)*128;
  out[i] = (1.f - fr)*p[xa] + fr*p[xb];
}

__global__ void k_convt3(const float* __restrict__ x, const float* __restrict__ w,
                         const float* __restrict__ bias, float* __restrict__ y,
                         int B, int Hh, int Wd){
  int i = blockIdx.x*256 + threadIdx.x;
  if (i >= B*Hh*Wd) return;
  int xw = i % Wd; int rem = i / Wd; int h = rem % Hh; int b = rem / Hh;
  float acc = bias[0];
  for (int p = 0; p < 3; p++)
    for (int q = 0; q < 3; q++) {
      int hi = h + p - 1, wi = xw + q - 1;
      if (hi >= 0 && hi < Hh && wi >= 0 && wi < Wd)
        acc += x[((size_t)b*Hh + hi)*Wd + wi] * w[(2-p)*3 + (2-q)];
    }
  y[i] = acc;
}

__global__ void k_stats_all(const float* __restrict__ x, float* st, int n){
  __shared__ float ss[256], sq[256];
  float s = 0.f, q = 0.f;
  for (int i = threadIdx.x; i < n; i += 256) { float v = x[i]; s += v; q += v*v; }
  ss[threadIdx.x] = s; sq[threadIdx.x] = q; __syncthreads();
  for (int stp = 128; stp > 0; stp >>= 1) {
    if (threadIdx.x < stp) { ss[threadIdx.x] += ss[threadIdx.x+stp]; sq[threadIdx.x] += sq[threadIdx.x+stp]; }
    __syncthreads();
  }
  if (threadIdx.x == 0) {
    float m = ss[0]/n; float v = sq[0]/n - m*m;
    st[0] = m; st[1] = rsqrtf(v + 1e-5f);
  }
}

__global__ void k_bn_elu_all(const float* __restrict__ x, float* __restrict__ y,
                             const float* st, const float* g, const float* b, int n){
  int i = blockIdx.x*256 + threadIdx.x;
  if (i >= n) return;
  float v = (x[i] - st[0]) * st[1] * g[0] + b[0];
  y[i] = v > 0.f ? v : expm1f(v);
}

// ---------------------------------------------------------------------------
// Host orchestration
// ---------------------------------------------------------------------------
static inline int cdiv(int a, int b){ return (a + b - 1) / b; }
static inline int cdivs(size_t a, size_t b){ return (int)((a + b - 1) / b); }

extern "C" void kernel_launch(void* const* d_in, const int* in_sizes, int n_in,
                              void* d_out, int out_size, void* d_ws, size_t ws_size,
                              hipStream_t stream) {
  (void)in_sizes; (void)n_in; (void)out_size; (void)ws_size;
  const float* const* in = (const float* const*)d_in;
  float* dout = (float*)d_out;

  char* ws = (char*)d_ws;
  size_t off = 0;
  auto alloc = [&](size_t bytes) -> char* {
    char* p = ws + off; off = (off + bytes + 255) & ~(size_t)255; return p;
  };

  // ---- scratch buffers ----
  float* meanb = (float*)alloc(2048*4);
  float* rstdb = (float*)alloc(2048*4);
  float* xyz0  = (float*)alloc((size_t)8*2048*3*4);
  float* pts0  = (float*)alloc((size_t)8*2048*4*4);
  float* l1xyz = (float*)alloc((size_t)8*1024*3*4);
  float* l2xyz = (float*)alloc((size_t)8*128*3*4);
  int*   fidx  = (int*)  alloc((size_t)8*1024*4);
  int*   bidx  = (int*)  alloc((size_t)8*1024*64*4);
  float* fpsd  = (float*)alloc((size_t)8*2048*4);
  float* l1pts = (float*)alloc((size_t)8*1024*128*4);
  float* l2pts = (float*)alloc((size_t)8*128*256*4);
  float* l2n   = (float*)alloc((size_t)8*128*256*4);
  float* l1n   = (float*)alloc((size_t)8*1024*128*4);
  int*   knni  = (int*)  alloc((size_t)8*2048*3*4);
  float* knnw  = (float*)alloc((size_t)8*2048*3*4);
  float* asfF  = (float*)alloc((size_t)8*1024*4);
  float* c1b   = (float*)alloc((size_t)8*128*251*4);
  float* c2b   = (float*)alloc((size_t)8*256*120*4);
  float* seqb  = (float*)alloc((size_t)120*8*256*4);
  float* hf    = (float*)alloc((size_t)120*8*64*4);
  float* hb    = (float*)alloc((size_t)120*8*64*4);
  __bf16* recb = (__bf16*)alloc((size_t)960*128*2);
  float* embF  = (float*)alloc((size_t)960*32*4);
  float* feats = (float*)alloc(8*32*4);
  float* lg    = (float*)alloc(8*16*4);
  __bf16* zlat = (__bf16*)alloc(8*32*2);
  __bf16* zg   = (__bf16*)alloc(8*32*2);
  __bf16* mupad= (__bf16*)alloc(8*32*2);
  float* smallF= (float*)alloc((size_t)8*2048*4);
  __bf16* smallA=(__bf16*)alloc((size_t)8*2048*2);
  __bf16* smallB=(__bf16*)alloc((size_t)8*2048*2);
  float* coarse= (float*)alloc((size_t)8*1792*4);
  float* sigT1 = (float*)alloc((size_t)8*8*512*4);
  float* sigT2 = (float*)alloc((size_t)8*8*512*4);
  float* stat2 = (float*)alloc(2*4);
  // big arenas (largest user: SA1 = 524288 rows x 128 ch)
  __bf16* bfA  = (__bf16*)alloc((size_t)524288*128*2);
  __bf16* bfB  = (__bf16*)alloc((size_t)524288*128*2);
  float*  F    = (float*) alloc((size_t)524288*128*4);

  // ---- weight packing (f32 -> padded bf16) ----
  auto prepw = [&](const float* w, int N, int K, int Kp, int trans) -> __bf16* {
    __bf16* o = (__bf16*)alloc((size_t)N*Kp*2);
    k_prepw<<<cdivs((size_t)N*Kp, 256), 256, 0, stream>>>(w, o, N, K, Kp, trans);
    return o;
  };
  __bf16* Wsa1_0 = prepw(in[0],  64, 7,   32,  0);
  __bf16* Wsa1_1 = prepw(in[4],  64, 64,  64,  0);
  __bf16* Wsa1_2 = prepw(in[8],  128,64,  64,  0);
  __bf16* Wsa2_0 = prepw(in[12], 128,131, 160, 0);
  __bf16* Wsa2_1 = prepw(in[16], 128,128, 128, 0);
  __bf16* Wsa2_2 = prepw(in[20], 256,128, 128, 0);
  // sa3 MLP + fc11 are dead (outputs never consumed) -> skipped
  __bf16* Wfc12  = prepw(in[38], 1024,32, 32,  1);
  __bf16* Wfp3_0 = prepw(in[40], 256,1280,1280,0);
  __bf16* Wfp3_1 = prepw(in[44], 256,256, 256, 0);
  __bf16* Wfp2_0 = prepw(in[48], 256,384, 384, 0);
  __bf16* Wfp2_1 = prepw(in[52], 128,256, 256, 0);
  __bf16* Wfp1_0 = prepw(in[56], 128,128, 128, 0);
  __bf16* Wfp1_1 = prepw(in[60], 128,128, 128, 0);
  __bf16* Wfp1_2 = prepw(in[64], 128,128, 128, 0);
  __bf16* Wc1    = prepw(in[68], 128,128, 128, 0);
  __bf16* Wc2    = prepw(in[72], 3,  128, 128, 0);
  __bf16* Wm0    = prepw(in[74], 256,16,  32,  1);
  __bf16* Wm1    = prepw(in[76], 512,256, 256, 1);
  __bf16* Wm2    = prepw(in[78], 1024,512,512, 1);
  __bf16* Wm3    = prepw(in[80], 2048,1024,1024,1);
  __bf16* Wm4    = prepw(in[82], 1792,2048,2048,1);
  __bf16* Wd0    = prepw(in[84], 512,25,  32,  0);
  __bf16* Wd1    = prepw(in[86], 512,512, 512, 0);
  __bf16* Wd2    = prepw(in[88], 7,  512, 512, 0);
  __bf16* Wemb   = prepw(in[106],32, 128, 128, 1);
  __bf16* Wsf1   = prepw(in[108],512,16,  32,  1);
  __bf16* Wsf2   = prepw(in[110],1024,512,512, 1);

  auto gemm = [&](const __bf16* Aa, const __bf16* Ww, const float* bias,
                  float* Cc, int M, int N, int Kp) {
    if (N >= 128) {
      dim3 g(cdiv(M, 128), cdiv(N, 128));
      k_gemm_big<<<g, 256, 0, stream>>>(Aa, Ww, bias, Cc, M, N, Kp);
    } else {
      dim3 g(cdiv(M, TBM), cdiv(N, TBN));
      k_gemm<<<g, 256, 0, stream>>>(Aa, Ww, bias, Cc, M, N, Kp);
    }
  };
  auto bnrelu = [&](float* X, __bf16* Y, const float* g, const float* b,
                    int M, int N, int Np) {
    k_colstats<<<N, 256, 0, stream>>>(X, meanb, rstdb, M, N);
    k_bn_act<<<cdivs((size_t)M*Np, 256), 256, 0, stream>>>(X, Y, meanb, rstdb, g, b, M, N, Np, 0);
  };

  const float* partial = in[120];
  const float* signal  = in[121];

  // ================= CRNN =================
  k_conv1d<<<cdiv(8*128*251,256),256,0,stream>>>(signal, in[90], in[91], c1b, 8,8,512,128,251,16,2,2);
  k_bnstats_bct<<<128,256,0,stream>>>(c1b, meanb, rstdb, 8,128,251);
  k_bnact_bct<<<cdiv(8*128*251,256),256,0,stream>>>(c1b, meanb, rstdb, in[92], in[93], 8,128,251, 0.2f);
  k_conv1d<<<cdiv(8*256*120,256),256,0,stream>>>(c1b, in[94], in[95], c2b, 8,128,251,256,120,16,2,2);
  k_bnstats_bct<<<256,256,0,stream>>>(c2b, meanb, rstdb, 8,256,120);
  k_bnact_bct<<<cdiv(8*256*120,256),256,0,stream>>>(c2b, meanb, rstdb, in[96], in[97], 8,256,120, 0.2f);
  k_bct_to_tbc<<<cdiv(8*256*120,256),256,0,stream>>>(c2b, seqb, 8,256,120);
  k_lstm<<<1,256,0,stream>>>(seqb, in[98],  in[99],  in[100], in[101], hf, 120,8,256,64, 0);
  k_lstm<<<1,256,0,stream>>>(seqb, in[102], in[103], in[104], in[105], hb, 120,8,256,64, 1);
  k_rec_concat<<<cdiv(960*128,256),256,0,stream>>>(hf, hb, recb, 960, 64);
  gemm(recb, Wemb, in[107], embF, 960, 32, 128);
  k_timemax<<<cdiv(8*32,256)+1,256,0,stream>>>(embF, feats, 120, 8, 32);
  k_latent<<<1,256,0,stream>>>(feats, dout+153600, dout+153728, lg, zlat, zg, 8, 16);

  // ================= SA1 =================
  k_split_partial<<<cdiv(8*2048,256),256,0,stream>>>(partial, xyz0, pts0, 8, 2048);
  k_fps<<<8,256,0,stream>>>(xyz0, fidx, fpsd, 2048, 1024);
  k_gather3<<<cdiv(8*1024,256),256,0,stream>>>(xyz0, fidx, l1xyz, 8, 2048, 1024);
  k_ballq<<<cdiv(8*1024,256),256,0,stream>>>(xyz0, l1xyz, bidx, 8, 2048, 1024, 64, 0.04f);
  k_group<<<cdivs((size_t)8*1024*64*32,256),256,0,stream>>>(xyz0, pts0, l1xyz, bidx, bfA, 8,2048,1024,64,4,32);
  gemm(bfA, Wsa1_0, in[1], F, 524288, 64, 32);
  bnrelu(F, bfB, in[2], in[3], 524288, 64, 64);
  gemm(bfB, Wsa1_1, in[5], F, 524288, 64, 64);
  bnrelu(F, bfA, in[6], in[7], 524288, 64, 64);
  gemm(bfA, Wsa1_2, in[9], F, 524288, 128, 64);
  bnrelu(F, bfB, in[10], in[11], 524288, 128, 128);
  k_maxpool_k<<<cdiv(8192*128,256),256,0,stream>>>(F, l1pts, 8192, 64, 128);

  // ================= SA2 =================
  k_fps<<<8,256,0,stream>>>(l1xyz, fidx, fpsd, 1024, 128);
  k_gather3<<<cdiv(8*128,256),256,0,stream>>>(l1xyz, fidx, l2xyz, 8, 1024, 128);
  k_ballq<<<cdiv(8*128,256),256,0,stream>>>(l1xyz, l2xyz, bidx, 8, 1024, 128, 64, 0.16f);
  k_group<<<cdivs((size_t)8*128*64*160,256),256,0,stream>>>(l1xyz, l1pts, l2xyz, bidx, bfA, 8,1024,128,64,128,160);
  gemm(bfA, Wsa2_0, in[13], F, 65536, 128, 160);
  bnrelu(F, bfB, in[14], in[15], 65536, 128, 128);
  gemm(bfB, Wsa2_1, in[17], F, 65536, 128, 128);
  bnrelu(F, bfA, in[18], in[19], 65536, 128, 128);
  gemm(bfA, Wsa2_2, in[21], F, 65536, 256, 128);
  bnrelu(F, bfB, in[22], in[23], 65536, 256, 256);
  k_maxpool_k<<<cdiv(1024*256,256),256,0,stream>>>(F, l2pts, 1024, 64, 256);
  // SA3 + fc11 are dead code (never reach the outputs) -> eliminated

  // ================= fc12 -> asf =================
  gemm(zlat, Wfc12, in[39], asfF, 8, 1024, 32);
  k_act<<<cdiv(8*1024,256),256,0,stream>>>(asfF, (__bf16*)nullptr, 8, 1024, 1024, 0);

  // ================= FP3 (interp == broadcast asf) =================
  k_concat_bcast<<<cdivs((size_t)1024*1280,256),256,0,stream>>>(l2pts, asfF, bfA, 8,128,256,1024,1280);
  gemm(bfA, Wfp3_0, in[41], F, 1024, 256, 1280);
  bnrelu(F, bfB, in[42], in[43], 1024, 256, 256);
  gemm(bfB, Wfp3_1, in[45], l2n, 1024, 256, 256);
  bnrelu(l2n, (__bf16*)nullptr, in[46], in[47], 1024, 256, 256);

  // ================= FP2 =================
  k_knn3<<<cdiv(8*1024,256),256,0,stream>>>(l1xyz, l2xyz, knni, knnw, 8, 1024, 128);
  k_interp_concat<<<cdivs((size_t)8192*384,256),256,0,stream>>>(l1pts, l2n, knni, knnw, bfA, 8,1024,128,128,256,384);
  gemm(bfA, Wfp2_0, in[49], F, 8192, 256, 384);
  bnrelu(F, bfB, in[50], in[51], 8192, 256, 256);
  gemm(bfB, Wfp2_1, in[53], l1n, 8192, 128, 256);
  bnrelu(l1n, (__bf16*)nullptr, in[54], in[55], 8192, 128, 128);

  // ================= FP1 =================
  k_knn3<<<cdiv(8*2048,256),256,0,stream>>>(xyz0, l1xyz, knni, knnw, 8, 2048, 1024);
  k_interp_concat<<<cdivs((size_t)16384*128,256),256,0,stream>>>((const float*)nullptr, l1n, knni, knnw, bfA, 8,2048,1024,0,128,128);
  gemm(bfA, Wfp1_0, in[57], F, 16384, 128, 128);
  bnrelu(F, bfB, in[58], in[59], 16384, 128, 128);
  gemm(bfB, Wfp1_1, in[61], F, 16384, 128, 128);
  bnrelu(F, bfA, in[62], in[63], 16384, 128, 128);
  gemm(bfA, Wfp1_2, in[65], F, 16384, 128, 128);
  bnrelu(F, bfB, in[66], in[67], 16384, 128, 128);

  // ================= segmentation head =================
  gemm(bfB, Wc1, in[69], F, 16384, 128, 128);
  bnrelu(F, bfA, in[70], in[71], 16384, 128, 128);
  gemm(bfA, Wc2, in[73], F, 16384, 3, 128);
  k_softmax_seg<<<cdiv(16384,256),256,0,stream>>>(F, dout, 8, 2048);          // y_seg @ 0

  // ================= geometry decoder (coarse MLP) =================
  gemm(zg, Wm0, in[75], smallF, 8, 256, 32);
  k_act<<<cdiv(8*256,256),256,0,stream>>>(smallF, smallA, 8, 256, 256, 0);
  gemm(smallA, Wm1, in[77], smallF, 8, 512, 256);
  k_act<<<cdiv(8*512,256),256,0,stream>>>(smallF, smallB, 8, 512, 512, 0);
  gemm(smallB, Wm2, in[79], smallF, 8, 1024, 512);
  k_act<<<cdiv(8*1024,256),256,0,stream>>>(smallF, smallA, 8, 1024, 1024, 0);
  gemm(smallA, Wm3, in[81], smallF, 8, 2048, 1024);
  k_act<<<cdiv(8*2048,256),256,0,stream>>>(smallF, smallB, 8, 2048, 2048, 0);
  gemm(smallB, Wm4, in[83], coarse, 8, 1792, 2048);
  k_sigmoid<<<cdiv(8*1792,256),256,0,stream>>>(coarse, dout+49152, 8*1792);   // y_coarse

  // ================= geometry decoder (fine path) =================
  k_dec_feat<<<cdivs((size_t)8192*32,256),256,0,stream>>>(coarse, lg, bfA, 8);
  gemm(bfA, Wd0, in[85], F, 8192, 512, 32);
  k_act<<<cdivs((size_t)8192*512,256),256,0,stream>>>(F, bfB, 8192, 512, 512, 0);
  gemm(bfB, Wd1, in[87], F, 8192, 512, 512);
  k_act<<<cdivs((size_t)8192*512,256),256,0,stream>>>(F, bfA, 8192, 512, 512, 0);
  gemm(bfA, Wd2, in[89], F, 8192, 7, 512);
  k_detail_out<<<cdiv(8*1024*7,256),256,0,stream>>>(F, coarse, dout+63488, 8); // y_detail

  // ================= signal decoder =================
  k_pad_mu<<<1,256,0,stream>>>(feats, mupad);
  gemm(mupad, Wsf1, in[109], smallF, 8, 512, 32);
  k_act<<<cdiv(8*512,256),256,0,stream>>>(smallF, smallB, 8, 512, 512, 1);     // elu
  gemm(smallB, Wsf2, in[111], smallF, 8, 1024, 512);
  k_act<<<cdiv(8*1024,256),256,0,stream>>>(smallF, (__bf16*)nullptr, 8, 1024, 1024, 1);
  k_resize<<<cdiv(8*8*512,256),256,0,stream>>>(smallF, sigT1, 8);
  k_convt3<<<cdiv(8*8*512,256),256,0,stream>>>(sigT1, in[112], in[113], sigT2, 8, 8, 512);
  k_stats_all<<<1,256,0,stream>>>(sigT2, stat2, 8*8*512);
  k_bn_elu_all<<<cdiv(32768,256),256,0,stream>>>(sigT2, sigT1, stat2, in[114], in[115], 32768);
  k_convt3<<<cdiv(8*8*512,256),256,0,stream>>>(sigT1, in[116], in[117], sigT2, 8, 8, 512);
  k_stats_all<<<1,256,0,stream>>>(sigT2, stat2, 8*8*512);
  k_bn_elu_all<<<cdiv(32768,256),256,0,stream>>>(sigT2, dout+120832, stat2, in[118], in[119], 32768); // y_ecg
  // mu_s @ 153600, std_s @ 153728 written by k_latent
}
